// MistralMHA_18150531793372
// MI455X (gfx1250) — matrix operand order repaired and checked by reading, not compiled, not run on hardware
//
#include <hip/hip_runtime.h>

// ---------------------------------------------------------------------------
// MI455X (gfx1250) Mistral-MHA forward:
//   [optional f32->f16 pre-conversion of x / wqkv / wo into workspace]
//   fused QKV+RoPE GEMM -> flash attention -> output projection.
// All matmuls via v_wmma_f32_16x16x32_f16 (wave32, f32 accumulate).
// ---------------------------------------------------------------------------

#define S_LEN 2048
#define DIM   4096
#define NH    32
#define NKV   8
#define HD    128
#define NEG_BIG (-1.0e9f)

typedef __attribute__((ext_vector_type(16))) _Float16 v16h;
typedef __attribute__((ext_vector_type(8)))  _Float16 v8h;
typedef __attribute__((ext_vector_type(8)))  float    v8f;
typedef __attribute__((ext_vector_type(4)))  float    v4f;

__device__ __forceinline__ v8f wmma16(v16h a, v16h b, v8f c) {
    return __builtin_amdgcn_wmma_f32_16x16x32_f16(
        /*neg_a=*/false, a, /*neg_b=*/false, b,
        /*c_mod=*/(short)0, c, /*reuse_a=*/false, /*reuse_b=*/false);
}

// ---- Fragment loaders (overloaded on source precision) --------------------
// A-matrix 16x32 f16 fragment (ISA 7.12.2): lanes 0-15 hold K {0..7,16..23},
// lanes 16-31 hold K {8..15,24..31} for row M = lane%16.
__device__ __forceinline__ v16h load_afrag(const _Float16* rowptr, int k, int lane) {
    int off = k + ((lane & 16) ? 8 : 0);
    v8h lo = *(const v8h*)(rowptr + off);
    v8h hi = *(const v8h*)(rowptr + off + 16);
    v16h r;
#pragma unroll
    for (int i = 0; i < 8; ++i) { r[i] = lo[i]; r[i + 8] = hi[i]; }
    return r;
}

__device__ __forceinline__ v16h load_afrag(const float* rowptr, int k, int lane) {
    int off = k + ((lane & 16) ? 8 : 0);
    v4f a = *(const v4f*)(rowptr + off);
    v4f b = *(const v4f*)(rowptr + off + 4);
    v4f c = *(const v4f*)(rowptr + off + 16);
    v4f d = *(const v4f*)(rowptr + off + 20);
    v16h r;
#pragma unroll
    for (int i = 0; i < 4; ++i) {
        r[i]      = (_Float16)a[i];
        r[i + 4]  = (_Float16)b[i];
        r[i + 8]  = (_Float16)c[i];
        r[i + 12] = (_Float16)d[i];
    }
    return r;
}

// B-matrix 32x16 f16 fragment: lane holds column N = lane%16; lanes 0-15 hold
// K {0..7,16..23}, lanes 16-31 hold K {8..15,24..31}.
__device__ __forceinline__ v16h load_bfrag(const _Float16* colptr, int k, int lane) {
    int off = k + ((lane & 16) ? 8 : 0);
    v8h lo = *(const v8h*)(colptr + off);
    v8h hi = *(const v8h*)(colptr + off + 16);
    v16h r;
#pragma unroll
    for (int i = 0; i < 8; ++i) { r[i] = lo[i]; r[i + 8] = hi[i]; }
    return r;
}

__device__ __forceinline__ v16h load_bfrag(const float* colptr, int k, int lane) {
    int off = k + ((lane & 16) ? 8 : 0);
    v4f a = *(const v4f*)(colptr + off);
    v4f b = *(const v4f*)(colptr + off + 4);
    v4f c = *(const v4f*)(colptr + off + 16);
    v4f d = *(const v4f*)(colptr + off + 20);
    v16h r;
#pragma unroll
    for (int i = 0; i < 4; ++i) {
        r[i]      = (_Float16)a[i];
        r[i + 4]  = (_Float16)b[i];
        r[i + 8]  = (_Float16)c[i];
        r[i + 12] = (_Float16)d[i];
    }
    return r;
}

template <typename T>
__device__ __forceinline__ const T* qkv_row(const T* wq, const T* wk, const T* wv, int n) {
    if (n < NH * HD)            return wq + (size_t)n * DIM;
    if (n < NH * HD + NKV * HD) return wk + (size_t)(n - NH * HD) * DIM;
    return wv + (size_t)(n - NH * HD - NKV * HD) * DIM;
}

// ---------------------------------------------------------------------------
// Kernel 0: bulk f32 -> f16 conversion (8 elems / thread, b128 in, b128 out).
// ---------------------------------------------------------------------------
__global__ __launch_bounds__(256) void cvt_f32_f16_kernel(
    const float* __restrict__ src, _Float16* __restrict__ dst, long n) {
    long i = ((long)blockIdx.x * 256 + threadIdx.x) * 8;
    if (i + 8 <= n) {
        v4f a = *(const v4f*)(src + i);
        v4f b = *(const v4f*)(src + i + 4);
        v8h r;
#pragma unroll
        for (int j = 0; j < 4; ++j) { r[j] = (_Float16)a[j]; r[j + 4] = (_Float16)b[j]; }
        *(v8h*)(dst + i) = r;
    }
}

// ---------------------------------------------------------------------------
// Kernel 1: C[2048, 6144] = x @ [wq|wk|wv]^T, fused RoPE, f16 outputs.
//   Q -> q_ws[h][s][d], K -> k_ws[hk][s][d], V -> v_ws[hk][d][s] (transposed).
// Block: 128 threads = 4 waves, 64x128 tile; each wave a 32x64 quadrant
// (2 A-frags x 4 B-frags -> 8 WMMAs per k-step).
// ---------------------------------------------------------------------------
template <typename TA, typename TB>
__global__ __launch_bounds__(128) void qkv_rope_kernel(
    const TA* __restrict__ x,
    const TB* __restrict__ wq, const TB* __restrict__ wk,
    const TB* __restrict__ wv,
    const float* __restrict__ fcos, const float* __restrict__ fsin,
    _Float16* __restrict__ q_ws, _Float16* __restrict__ k_ws,
    _Float16* __restrict__ v_ws) {

    __shared__ float tile[64 * 128];

    const int tid  = threadIdx.x;
    const int wave = tid >> 5;
    const int lane = tid & 31;
    const int nbase = blockIdx.x * 128;
    const int mbase = blockIdx.y * 64;
    const int wr = (wave >> 1) * 32;   // wave row offset in tile
    const int wc = (wave & 1) * 64;    // wave col offset in tile

    const TA* arow0 = x + (size_t)(mbase + wr + (lane & 15)) * DIM;
    const TA* arow1 = arow0 + (size_t)16 * DIM;
    const TB* brow[4];
#pragma unroll
    for (int tj = 0; tj < 4; ++tj)
        brow[tj] = qkv_row(wq, wk, wv, nbase + wc + tj * 16 + (lane & 15));

    v8f acc[2][4];
#pragma unroll
    for (int ti = 0; ti < 2; ++ti)
#pragma unroll
        for (int tj = 0; tj < 4; ++tj) acc[ti][tj] = (v8f){};

    for (int k = 0; k < DIM; k += 32) {
        __builtin_prefetch(brow[0] + k + 256, 0, 0);
        v16h a0 = load_afrag(arow0, k, lane);
        v16h a1 = load_afrag(arow1, k, lane);
#pragma unroll
        for (int tj = 0; tj < 4; ++tj) {
            v16h b = load_bfrag(brow[tj], k, lane);
            acc[0][tj] = wmma16(a0, b, acc[0][tj]);
            acc[1][tj] = wmma16(a1, b, acc[1][tj]);
        }
    }

    // Stage f32 tile through LDS so RoPE pairs (adjacent columns) meet in one lane.
    const int rh = (lane & 16) ? 8 : 0;
#pragma unroll
    for (int ti = 0; ti < 2; ++ti)
#pragma unroll
        for (int tj = 0; tj < 4; ++tj)
#pragma unroll
            for (int g = 0; g < 8; ++g)
                tile[(wr + ti * 16 + g + rh) * 128 + wc + tj * 16 + (lane & 15)] =
                    acc[ti][tj][g];
    __syncthreads();

#pragma unroll
    for (int i = 0; i < 32; ++i) {
        int p   = i * 128 + tid;      // 0..4095 pairs in the 64x128 tile
        int row = p >> 6;
        int nl  = (p & 63) * 2;
        int m   = mbase + row;
        int n   = nbase + nl;
        float re = tile[row * 128 + nl];
        float im = tile[row * 128 + nl + 1];
        if (n < (NH + NKV) * HD) {
            // Q or K: rotary embedding on the (re, im) pair.
            int d  = n & (HD - 1);
            int fi = d >> 1;
            float c = fcos[m * (HD / 2) + fi];
            float s = fsin[m * (HD / 2) + fi];
            float ore = re * c - im * s;
            float oim = re * s + im * c;
            if (n < NH * HD) {
                int hq = n >> 7;
                _Float16* dst = q_ws + ((size_t)hq * S_LEN + m) * HD + d;
                dst[0] = (_Float16)ore;
                dst[1] = (_Float16)oim;
            } else {
                int nk = n - NH * HD;
                int hk = nk >> 7;
                _Float16* dst = k_ws + ((size_t)hk * S_LEN + m) * HD + (nk & (HD - 1));
                dst[0] = (_Float16)ore;
                dst[1] = (_Float16)oim;
            }
        } else {
            // V: no RoPE, store transposed [hk][d][s].
            int nv = n - (NH + NKV) * HD;
            int hv = nv >> 7;
            int d  = nv & (HD - 1);
            v_ws[((size_t)hv * HD + d) * S_LEN + m]     = (_Float16)re;
            v_ws[((size_t)hv * HD + d + 1) * S_LEN + m] = (_Float16)im;
        }
    }
}

// ---------------------------------------------------------------------------
// Kernel 2: flash attention.  Grid (32 q-tiles, 32 heads), 4 waves/block,
// each wave owns 16 query rows, streams KV in 32-wide chunks with online
// softmax.  P is transposed C-layout -> A-layout through per-wave LDS.
// ---------------------------------------------------------------------------
__global__ __launch_bounds__(128) void flash_attn_kernel(
    const _Float16* __restrict__ q_ws, const _Float16* __restrict__ k_ws,
    const _Float16* __restrict__ v_ws, _Float16* __restrict__ o_ws) {

    __shared__ _Float16 plds[4 * 16 * 32];   // per-wave 16x32 f16 P tile

    const int tid  = threadIdx.x;
    const int wave = tid >> 5;
    const int lane = tid & 31;
    const int h    = blockIdx.y;
    const int hk   = h >> 2;                   // REPEATS = 4
    const int qrow = blockIdx.x * 64 + wave * 16;
    const int rh   = (lane & 16) ? 8 : 0;
    const float scale = 0.088388347648318447f;  // 1/sqrt(128)

    // Preload the wave's Q A-fragments (16 rows x 128 dims = 4 fragments).
    const _Float16* qptr = q_ws + ((size_t)h * S_LEN + qrow + (lane & 15)) * HD;
    v16h aQ[4];
#pragma unroll
    for (int f = 0; f < 4; ++f) aQ[f] = load_afrag(qptr, 32 * f, lane);

    v8f oC[8];
#pragma unroll
    for (int dj = 0; dj < 8; ++dj) oC[dj] = (v8f){};
    float mrow[8], lrow[8];
#pragma unroll
    for (int g = 0; g < 8; ++g) { mrow[g] = NEG_BIG; lrow[g] = 0.0f; }

    _Float16* pl = plds + wave * (16 * 32);

    for (int jb = 0; jb <= qrow + 15; jb += 32) {
        v8f s[2];
#pragma unroll
        for (int sub = 0; sub < 2; ++sub) {
            const int kvb = jb + sub * 16;
            s[sub] = (v8f){};
            if (kvb <= qrow + 15) {
                const _Float16* kptr =
                    k_ws + ((size_t)hk * S_LEN + kvb + (lane & 15)) * HD;
#pragma unroll
                for (int f = 0; f < 4; ++f)
                    s[sub] = wmma16(aQ[f], load_bfrag(kptr, 32 * f, lane), s[sub]);
                const int col = kvb + (lane & 15);
#pragma unroll
                for (int g = 0; g < 8; ++g) {
                    int row = qrow + g + rh;
                    s[sub][g] = (col <= row) ? s[sub][g] * scale : NEG_BIG;
                }
            } else {
#pragma unroll
                for (int g = 0; g < 8; ++g) s[sub][g] = NEG_BIG;
            }
        }

        // Online softmax: row stats reduced across the 16 lanes of each half.
#pragma unroll
        for (int g = 0; g < 8; ++g) {
            float mx = fmaxf(s[0][g], s[1][g]);
            mx = fmaxf(mx, __shfl_xor(mx, 1, 32));
            mx = fmaxf(mx, __shfl_xor(mx, 2, 32));
            mx = fmaxf(mx, __shfl_xor(mx, 4, 32));
            mx = fmaxf(mx, __shfl_xor(mx, 8, 32));
            float mnew = fmaxf(mrow[g], mx);
            float corr = __expf(mrow[g] - mnew);
            float p0 = __expf(s[0][g] - mnew);
            float p1 = __expf(s[1][g] - mnew);
            float rs = p0 + p1;
            rs += __shfl_xor(rs, 1, 32);
            rs += __shfl_xor(rs, 2, 32);
            rs += __shfl_xor(rs, 4, 32);
            rs += __shfl_xor(rs, 8, 32);
            lrow[g] = lrow[g] * corr + rs;
            mrow[g] = mnew;
            s[0][g] = p0;
            s[1][g] = p1;
#pragma unroll
            for (int dj = 0; dj < 8; ++dj) oC[dj][g] *= corr;
        }

        // C-layout P -> LDS (f16, 16x32 row-major) -> A-fragment.
#pragma unroll
        for (int g = 0; g < 8; ++g) {
            pl[(g + rh) * 32 + (lane & 15)]      = (_Float16)s[0][g];
            pl[(g + rh) * 32 + 16 + (lane & 15)] = (_Float16)s[1][g];
        }
        asm volatile("s_wait_dscnt 0" ::: "memory");   // intra-wave LDS RAW
        v16h aP = load_afrag(pl + (lane & 15) * 32, 0, lane);
        asm volatile("s_wait_dscnt 0" ::: "memory");   // keep next stores after loads

        // P (16x32) @ V (32x128): 8 WMMAs, V fragments contiguous ([hk][d][s]).
#pragma unroll
        for (int dj = 0; dj < 8; ++dj) {
            const _Float16* vptr =
                v_ws + ((size_t)hk * HD + dj * 16 + (lane & 15)) * S_LEN;
            oC[dj] = wmma16(aP, load_bfrag(vptr, jb, lane), oC[dj]);
        }
    }

    // Normalize and emit f16 rows: o_ws[m][h*128 + d], K-contiguous for O-proj.
#pragma unroll
    for (int g = 0; g < 8; ++g) {
        float inv = 1.0f / lrow[g];
        int m = qrow + g + rh;
        _Float16* dst = o_ws + (size_t)m * (NH * HD) + h * HD + (lane & 15);
#pragma unroll
        for (int dj = 0; dj < 8; ++dj)
            dst[dj * 16] = (_Float16)(oC[dj][g] * inv);
    }
}

// ---------------------------------------------------------------------------
// Kernel 3: out[2048,4096] = o_ws (f16) @ wo rows, f32 output.
// Block: 128 threads = 4 waves, 64x128 tile; wave = 32x64 (8 WMMAs / k-step).
// ---------------------------------------------------------------------------
template <typename TB>
__global__ __launch_bounds__(128) void oproj_kernel(
    const _Float16* __restrict__ o_ws, const TB* __restrict__ wo,
    float* __restrict__ out) {

    const int tid  = threadIdx.x;
    const int wave = tid >> 5;
    const int lane = tid & 31;
    const int nbase = blockIdx.x * 128;
    const int mbase = blockIdx.y * 64;
    const int wr = (wave >> 1) * 32;
    const int wc = (wave & 1) * 64;
    const int rh = (lane & 16) ? 8 : 0;

    const _Float16* arow0 = o_ws + (size_t)(mbase + wr + (lane & 15)) * DIM;
    const _Float16* arow1 = arow0 + (size_t)16 * DIM;
    const TB* brow[4];
#pragma unroll
    for (int tj = 0; tj < 4; ++tj)
        brow[tj] = wo + (size_t)(nbase + wc + tj * 16 + (lane & 15)) * DIM;

    v8f acc[2][4];
#pragma unroll
    for (int ti = 0; ti < 2; ++ti)
#pragma unroll
        for (int tj = 0; tj < 4; ++tj) acc[ti][tj] = (v8f){};

    for (int k = 0; k < DIM; k += 32) {
        __builtin_prefetch(brow[0] + k + 256, 0, 0);
        v16h a0 = load_afrag(arow0, k, lane);
        v16h a1 = load_afrag(arow1, k, lane);
#pragma unroll
        for (int tj = 0; tj < 4; ++tj) {
            v16h b = load_bfrag(brow[tj], k, lane);
            acc[0][tj] = wmma16(a0, b, acc[0][tj]);
            acc[1][tj] = wmma16(a1, b, acc[1][tj]);
        }
    }

#pragma unroll
    for (int ti = 0; ti < 2; ++ti)
#pragma unroll
        for (int tj = 0; tj < 4; ++tj)
#pragma unroll
            for (int g = 0; g < 8; ++g) {
                int m0 = mbase + wr + ti * 16 + g + rh;
                int c0 = nbase + wc + tj * 16 + (lane & 15);
                out[(size_t)m0 * DIM + c0] = acc[ti][tj][g];
            }
}

// ---------------------------------------------------------------------------
extern "C" void kernel_launch(void* const* d_in, const int* in_sizes, int n_in,
                              void* d_out, int out_size, void* d_ws, size_t ws_size,
                              hipStream_t stream) {
    (void)in_sizes; (void)n_in; (void)out_size;

    const float* x    = (const float*)d_in[0];
    const float* fcos = (const float*)d_in[1];
    const float* fsin = (const float*)d_in[2];
    // d_in[3] = mask (unused: window == S -> pure causal, applied analytically)
    // d_in[4] = positions (unused: positions == arange(S))
    const float* wq = (const float*)d_in[5];
    const float* wk = (const float*)d_in[6];
    const float* wv = (const float*)d_in[7];
    const float* wo = (const float*)d_in[8];
    float* out = (float*)d_out;

    const size_t MB = 1024 * 1024;
    char* ws = (char*)d_ws;
    _Float16* q_ws = (_Float16*)(ws);               // 16 MiB  [h][s][d]
    _Float16* k_ws = (_Float16*)(ws + 16 * MB);     //  4 MiB  [hk][s][d]
    _Float16* v_ws = (_Float16*)(ws + 20 * MB);     //  4 MiB  [hk][d][s]
    _Float16* o_ws = (_Float16*)(ws + 24 * MB);     // 16 MiB  [s][h*d]
    _Float16* x16  = (_Float16*)(ws + 40 * MB);     // 16 MiB
    _Float16* w16  = (_Float16*)(ws + 56 * MB);     // 48 MiB  (wq|wk|wv rows)
    _Float16* wo16 = (_Float16*)(ws + 104 * MB);    // 32 MiB
    const bool pre_convert = ws_size >= 136 * MB;   // fixed per run -> deterministic

    if (pre_convert) {
        // One bandwidth-bound pass so the GEMM hot loops are pure f16 b128 loads.
        cvt_f32_f16_kernel<<<(long)S_LEN * DIM / (256 * 8), 256, 0, stream>>>(
            x, x16, (long)S_LEN * DIM);
        cvt_f32_f16_kernel<<<(long)NH * HD * DIM / (256 * 8), 256, 0, stream>>>(
            wq, w16, (long)NH * HD * DIM);
        cvt_f32_f16_kernel<<<(long)NKV * HD * DIM / (256 * 8), 256, 0, stream>>>(
            wk, w16 + (size_t)NH * HD * DIM, (long)NKV * HD * DIM);
        cvt_f32_f16_kernel<<<(long)NKV * HD * DIM / (256 * 8), 256, 0, stream>>>(
            wv, w16 + (size_t)(NH + NKV) * HD * DIM, (long)NKV * HD * DIM);
        cvt_f32_f16_kernel<<<(long)DIM * DIM / (256 * 8), 256, 0, stream>>>(
            wo, wo16, (long)DIM * DIM);

        qkv_rope_kernel<_Float16, _Float16>
            <<<dim3((NH + 2 * NKV) * HD / 128, S_LEN / 64), 128, 0, stream>>>(
                x16, w16, w16 + (size_t)NH * HD * DIM,
                w16 + (size_t)(NH + NKV) * HD * DIM,
                fcos, fsin, q_ws, k_ws, v_ws);
    } else {
        qkv_rope_kernel<float, float>
            <<<dim3((NH + 2 * NKV) * HD / 128, S_LEN / 64), 128, 0, stream>>>(
                x, wq, wk, wv, fcos, fsin, q_ws, k_ws, v_ws);
    }

    // Flash attention (32 q-tiles of 64 rows x 32 heads).
    flash_attn_kernel<<<dim3(S_LEN / 64, NH), 128, 0, stream>>>(
        q_ws, k_ws, v_ws, o_ws);

    // Output projection (4096 cols x 2048 rows, 64x128 tiles).
    if (pre_convert) {
        oproj_kernel<_Float16><<<dim3(DIM / 128, S_LEN / 64), 128, 0, stream>>>(
            o_ws, wo16, out);
    } else {
        oproj_kernel<float><<<dim3(DIM / 128, S_LEN / 64), 128, 0, stream>>>(
            o_ws, wo, out);
    }
}
